// BertSelfAttention_81003083203093
// MI455X (gfx1250) — matrix operand order repaired and checked by reading, not compiled, not run on hardware
//
#include <hip/hip_runtime.h>

// ---------------------------------------------------------------------------
// Problem constants (from reference): B=8, S=512, Hid=1024, H=16, D=64
// ---------------------------------------------------------------------------
#define BB   8
#define SS   512
#define HID  1024
#define NH   16
#define DH   64

typedef __attribute__((ext_vector_type(16))) __bf16 v16bf;
typedef __attribute__((ext_vector_type(8)))  __bf16 v8bf;
typedef __attribute__((ext_vector_type(8)))  float  v8f;

union BF16x16 { v16bf v; v8bf h[2]; };

// ---------------------------------------------------------------------------
// Kernel 1: f32 -> bf16 conversion, float4 per thread (n must be mult. of 4)
// ---------------------------------------------------------------------------
__global__ void cvt_f32_bf16(const float* __restrict__ src,
                             unsigned short* __restrict__ dst, int n4) {
  const int i = blockIdx.x * blockDim.x + threadIdx.x;
  if (i >= n4) return;
  const float4 x = ((const float4*)src)[i];
  __bf16* d = (__bf16*)dst + 4 * (size_t)i;
  d[0] = (__bf16)x.x; d[1] = (__bf16)x.y;
  d[2] = (__bf16)x.z; d[3] = (__bf16)x.w;
}

// ---------------------------------------------------------------------------
// Kernel 2: QKV projection GEMM.  y[m,n] = sum_k X[m,k] * W[n,k] + bias[n]
//   One wave computes a 64x64 tile: 4 A-frags x 4 B-frags -> 16 WMMA/k-step,
//   4x operand reuse vs a 16x64 tile (projection is L2-bandwidth bound).
//   which = blockIdx.y: 0->Q ([B,H,S,D]), 1->K ([B,H,S,D]), 2->V^T ([B,H,D,S])
// ---------------------------------------------------------------------------
__global__ __launch_bounds__(32)
void qkv_gemm(const unsigned short* __restrict__ Xb,
              const unsigned short* __restrict__ Wb,
              const float* __restrict__ bq, const float* __restrict__ bk,
              const float* __restrict__ bv,
              unsigned short* __restrict__ Qo,
              unsigned short* __restrict__ Ko,
              unsigned short* __restrict__ Vt) {
  const int lane  = threadIdx.x;
  const int which = blockIdx.y;
  const int m0 = (blockIdx.x & 63) * 64;    // 4096/64 = 64 row tiles
  const int n0 = (blockIdx.x >> 6) * 64;    // 1024/64 = 16 col tiles

  const __bf16* X = (const __bf16*)Xb;
  const __bf16* W = (const __bf16*)Wb + (size_t)which * HID * HID;
  const float* bias = (which == 0) ? bq : (which == 1) ? bk : bv;

  v8f c[4][4] = {};                          // c[am][bn]
  const int akb = (lane >> 4) << 4;          // A: +16 K for upper half-wave
  const int bkb = (lane >> 4) << 4;          // B: +16 K for upper half-wave

  for (int k0 = 0; k0 < HID; k0 += 32) {
    BF16x16 a[4];
#pragma unroll
    for (int am = 0; am < 4; ++am) {
      const __bf16* xr =
          X + (size_t)(m0 + am * 16 + (lane & 15)) * HID + k0 + akb;
      a[am].h[0] = *(const v8bf*)(xr);       // K = k0+akb .. +7
      a[am].h[1] = *(const v8bf*)(xr + 8);   // K = k0+akb+8 .. +15
    }
#pragma unroll
    for (int bn = 0; bn < 4; ++bn) {
      const int wrow = n0 + bn * 16 + (lane & 15);
      const v16bf bm = *(const v16bf*)(W + (size_t)wrow * HID + k0 + bkb);
#pragma unroll
      for (int am = 0; am < 4; ++am)
        c[am][bn] = __builtin_amdgcn_wmma_f32_16x16x32_bf16(
            false, a[am].v, false, bm, (short)0, c[am][bn], false, false);
    }
  }

  // C/D layout: lane<16 -> M=j, lane>=16 -> M=8+j ; N = lane&15
#pragma unroll
  for (int am = 0; am < 4; ++am) {
#pragma unroll
    for (int bn = 0; bn < 4; ++bn) {
#pragma unroll
      for (int j = 0; j < 8; ++j) {
        const int m = m0 + am * 16 + j + ((lane >> 4) << 3);
        const int n = n0 + bn * 16 + (lane & 15);
        const float val = c[am][bn][j] + bias[n];
        const int b = m >> 9, s = m & (SS - 1);
        const int h = n >> 6, d = n & (DH - 1);
        if (which == 0)
          ((__bf16*)Qo)[(((size_t)(b * NH + h) * SS) + s) * DH + d] = (__bf16)val;
        else if (which == 1)
          ((__bf16*)Ko)[(((size_t)(b * NH + h) * SS) + s) * DH + d] = (__bf16)val;
        else
          ((__bf16*)Vt)[(((size_t)(b * NH + h) * DH) + d) * SS + s] = (__bf16)val;
      }
    }
  }
}

// ---------------------------------------------------------------------------
// Kernel 3: relation score table  t[b,h,s,r-1] = sum_d Q[b,h,s,d]*dp_k[r,d]
// ---------------------------------------------------------------------------
__global__ void rel_t_kernel(const unsigned short* __restrict__ Qb,
                             const float* __restrict__ dpk,
                             float* __restrict__ T) {
  const int i = blockIdx.x * blockDim.x + threadIdx.x;  // over B*H*S
  if (i >= BB * NH * SS) return;
  const __bf16* q = (const __bf16*)Qb + (size_t)i * DH;
  float a1 = 0.f, a2 = 0.f;
#pragma unroll 8
  for (int d = 0; d < DH; ++d) {
    const float qd = (float)q[d];
    a1 += qd * dpk[DH + d];
    a2 += qd * dpk[2 * DH + d];
  }
  T[2 * i + 0] = a1;
  T[2 * i + 1] = a2;
}

// ---------------------------------------------------------------------------
// Kernel 4: fused attention.  One wave handles (b, h, 16-query block).
//   K tiles are double-buffered into LDS via global_load_async_to_lds_b128,
//   software-pipelined with s_wait_asynccnt (in-order completion => waiting
//   for <=4 outstanding after issuing 4 new guarantees current buffer done).
// ---------------------------------------------------------------------------
#define ASYNC_K_STAGE(ldsoff, gptr)                                          \
  asm volatile("global_load_async_to_lds_b128 %0, %1, off\n\t"               \
               "global_load_async_to_lds_b128 %0, %1, off offset:512\n\t"    \
               "global_load_async_to_lds_b128 %0, %1, off offset:1024\n\t"   \
               "global_load_async_to_lds_b128 %0, %1, off offset:1536"       \
               :: "v"(ldsoff), "v"(gptr) : "memory")

__global__ __launch_bounds__(32)
void attn_kernel(const unsigned short* __restrict__ Qb,
                 const unsigned short* __restrict__ Kb,
                 const unsigned short* __restrict__ Vtb,
                 const float* __restrict__ T,
                 const int* __restrict__ ge,
                 const float* __restrict__ mask,
                 const float* __restrict__ dpv,
                 float* __restrict__ out) {
  __shared__ __align__(16) unsigned short kbuf[2 * 16 * DH];  // 4 KB, 2 bufs
  __shared__ __align__(16) float sc[16 * SS];                 // 32 KB strip
  __shared__ float m1[16], m2[16];

  const int lane = threadIdx.x;
  const int wg = blockIdx.x;
  const int q0 = (wg & 31) * 16;
  const int h  = (wg >> 5) & (NH - 1);
  const int b  = wg >> 9;

  const __bf16* Q  = (const __bf16*)Qb  + (size_t)(b * NH + h) * SS * DH;
  const __bf16* K  = (const __bf16*)Kb  + (size_t)(b * NH + h) * SS * DH;
  const __bf16* Vt = (const __bf16*)Vtb + (size_t)(b * NH + h) * DH * SS;

  // LDS byte addresses (low 32 bits of generic pointer = LDS offset)
  const unsigned kbase = (unsigned)(size_t)(&kbuf[0]) + (unsigned)(lane * 16);

  // ---- Q A-fragments (two 16x32 chunks over d) ----
  BF16x16 aq[2];
  {
    const int row = q0 + (lane & 15);
    const int kb  = (lane >> 4) << 4;
    const __bf16* qr = Q + (size_t)row * DH;
    aq[0].h[0] = *(const v8bf*)(qr + kb);
    aq[0].h[1] = *(const v8bf*)(qr + kb + 8);
    aq[1].h[0] = *(const v8bf*)(qr + 32 + kb);
    aq[1].h[1] = *(const v8bf*)(qr + 32 + kb + 8);
  }

  // ---- scores: async-pipelined loop over 16-key tiles ----
  const float inv_sqrt_d = 0.125f;
  const int boff = (lane >> 4) << 4;
  ASYNC_K_STAGE(kbase, K + (size_t)lane * 8);   // prefetch tile 0 -> buf 0
  for (int n0 = 0; n0 < SS; n0 += 16) {
    const int buf = (n0 >> 4) & 1;
    if (n0 + 16 < SS) {
      ASYNC_K_STAGE(kbase + (unsigned)((buf ^ 1) * 2048),
                    K + (size_t)(n0 + 16) * DH + lane * 8);
      asm volatile("s_wait_asynccnt 0x4" ::: "memory");
    } else {
      asm volatile("s_wait_asynccnt 0x0" ::: "memory");
    }
    const __bf16* kr = (const __bf16*)kbuf + buf * 1024 + (lane & 15) * DH;
    const v16bf b0 = *(const v16bf*)(kr + boff);
    const v16bf b1 = *(const v16bf*)(kr + 32 + boff);
    v8f c = {};
    c = __builtin_amdgcn_wmma_f32_16x16x32_bf16(false, aq[0].v, false, b0,
                                                (short)0, c, false, false);
    c = __builtin_amdgcn_wmma_f32_16x16x32_bf16(false, aq[1].v, false, b1,
                                                (short)0, c, false, false);
    const int kidx = n0 + (lane & 15);
    const float mk = mask[b * SS + kidx];
#pragma unroll
    for (int j = 0; j < 8; ++j) {
      const int ql = j + ((lane >> 4) << 3);
      const int qg = q0 + ql;
      const int g  = ge[((size_t)(b * SS + qg)) * SS + kidx];
      const float* tr = T + ((size_t)(b * NH + h) * SS + qg) * 2;
      const float add = (g == 1) ? tr[0] : (g == 2) ? tr[1] : 0.f;
      sc[ql * SS + kidx] = c[j] * inv_sqrt_d + add + mk;
    }
  }
  __syncthreads();

  // ---- softmax + relation masses: all 32 lanes, float4-vectorized.
  //      row = lane&15, half = lane>>4 (256 elems each), combine via xor-16.
  {
    const int srow = lane & 15;
    const int half = lane >> 4;
    float4* rowv = (float4*)(sc + srow * SS) + half * 64;
    float mx = -1e30f;
#pragma unroll 4
    for (int i = 0; i < 64; ++i) {
      const float4 x = rowv[i];
      mx = fmaxf(mx, fmaxf(fmaxf(x.x, x.y), fmaxf(x.z, x.w)));
    }
    mx = fmaxf(mx, __shfl_xor(mx, 16));
    float sum = 0.f;
#pragma unroll 4
    for (int i = 0; i < 64; ++i) {
      float4 x = rowv[i];
      x.x = __expf(x.x - mx); x.y = __expf(x.y - mx);
      x.z = __expf(x.z - mx); x.w = __expf(x.w - mx);
      sum += (x.x + x.y) + (x.z + x.w);
      rowv[i] = x;
    }
    sum += __shfl_xor(sum, 16);
    const float invs = 1.f / sum;
    const int4* gr =
        (const int4*)(ge + ((size_t)(b * SS + q0 + srow)) * SS) + half * 64;
    float a1 = 0.f, a2 = 0.f;
#pragma unroll 4
    for (int i = 0; i < 64; ++i) {
      float4 x = rowv[i];
      const int4 g = gr[i];
      x.x *= invs; x.y *= invs; x.z *= invs; x.w *= invs;
      if (g.x == 1) a1 += x.x; else if (g.x == 2) a2 += x.x;
      if (g.y == 1) a1 += x.y; else if (g.y == 2) a2 += x.y;
      if (g.z == 1) a1 += x.z; else if (g.z == 2) a2 += x.z;
      if (g.w == 1) a1 += x.w; else if (g.w == 2) a2 += x.w;
      rowv[i] = x;
    }
    a1 += __shfl_xor(a1, 16);
    a2 += __shfl_xor(a2, 16);
    if (half == 0) { m1[srow] = a1; m2[srow] = a2; }
  }
  __syncthreads();

  // ---- context: P[16x512] (LDS -> bf16) x V^T tiles ----
  v8f cacc[4] = {};
  for (int k0 = 0; k0 < SS; k0 += 32) {
    BF16x16 ap;
    const float* pr = sc + (lane & 15) * SS + k0 + ((lane >> 4) << 4);
#pragma unroll
    for (int e2 = 0; e2 < 4; ++e2) {
      const float2 x = ((const float2*)pr)[e2];        // K = kb+2e2 .. +1
      const float2 y = ((const float2*)(pr + 8))[e2];  // K = kb+8+2e2 .. +1
      ap.v[2 * e2]     = (__bf16)x.x;
      ap.v[2 * e2 + 1] = (__bf16)x.y;
      ap.v[8 + 2 * e2]     = (__bf16)y.x;
      ap.v[8 + 2 * e2 + 1] = (__bf16)y.y;
    }
#pragma unroll
    for (int t = 0; t < 4; ++t) {
      const int d = t * 16 + (lane & 15);
      const v16bf bv = *(const v16bf*)(Vt + (size_t)d * SS + k0 + boff);
      cacc[t] = __builtin_amdgcn_wmma_f32_16x16x32_bf16(
          false, ap.v, false, bv, (short)0, cacc[t], false, false);
    }
  }

  // ---- epilogue: + relation-value term, write [B,S,H*D] ----
#pragma unroll
  for (int t = 0; t < 4; ++t) {
#pragma unroll
    for (int j = 0; j < 8; ++j) {
      const int ql = j + ((lane >> 4) << 3);
      const int d  = t * 16 + (lane & 15);
      const float val =
          cacc[t][j] + m1[ql] * dpv[DH + d] + m2[ql] * dpv[2 * DH + d];
      out[((size_t)(b * SS + q0 + ql)) * (NH * DH) + h * DH + d] = val;
    }
  }
}

// ---------------------------------------------------------------------------
// Host launcher
// ---------------------------------------------------------------------------
extern "C" void kernel_launch(void* const* d_in, const int* in_sizes, int n_in,
                              void* d_out, int out_size, void* d_ws, size_t ws_size,
                              hipStream_t stream) {
  const float* hidden = (const float*)d_in[0];
  const float* amask  = (const float*)d_in[1];
  const int*   gemb   = (const int*)d_in[2];
  const float* Wq     = (const float*)d_in[3];
  const float* bq     = (const float*)d_in[4];
  const float* Wk     = (const float*)d_in[5];
  const float* bk     = (const float*)d_in[6];
  const float* Wv     = (const float*)d_in[7];
  const float* bv     = (const float*)d_in[8];
  const float* dpk    = (const float*)d_in[9];
  const float* dpv    = (const float*)d_in[10];
  float* out = (float*)d_out;

  // workspace carve-up (all 256B-aligned offsets)
  char* w = (char*)d_ws;
  unsigned short* Xbf = (unsigned short*)w; w += (size_t)BB * SS * HID * 2;      // 8 MB
  unsigned short* Wbf = (unsigned short*)w; w += (size_t)3 * HID * HID * 2;      // 6 MB
  unsigned short* Qbf = (unsigned short*)w; w += (size_t)BB * NH * SS * DH * 2;  // 8 MB
  unsigned short* Kbf = (unsigned short*)w; w += (size_t)BB * NH * SS * DH * 2;  // 8 MB
  unsigned short* Vt  = (unsigned short*)w; w += (size_t)BB * NH * DH * SS * 2;  // 8 MB
  float* T = (float*)w;                                                          // 512 KB

  cvt_f32_bf16<<<(BB * SS * HID / 4 + 255) / 256, 256, 0, stream>>>(
      hidden, Xbf, BB * SS * HID / 4);
  cvt_f32_bf16<<<(HID * HID / 4 + 255) / 256, 256, 0, stream>>>(
      Wq, Wbf, HID * HID / 4);
  cvt_f32_bf16<<<(HID * HID / 4 + 255) / 256, 256, 0, stream>>>(
      Wk, Wbf + (size_t)HID * HID, HID * HID / 4);
  cvt_f32_bf16<<<(HID * HID / 4 + 255) / 256, 256, 0, stream>>>(
      Wv, Wbf + (size_t)2 * HID * HID, HID * HID / 4);

  qkv_gemm<<<dim3(1024, 3), 32, 0, stream>>>(Xbf, Wbf, bq, bk, bv, Qbf, Kbf, Vt);

  rel_t_kernel<<<(BB * NH * SS + 255) / 256, 256, 0, stream>>>(Qbf, dpk, T);

  attn_kernel<<<BB * NH * (SS / 16), 32, 0, stream>>>(Qbf, Kbf, Vt, T, gemb,
                                                      amask, dpv, out);
}
